// VMamba2D_18124761989174
// MI455X (gfx1250) — matrix operand order repaired and checked by reading, not compiled, not run on hardware
//
#include <hip/hip_runtime.h>
#include <hip/hip_bf16.h>

#define DM   192      // D_MODEL
#define DI   384      // D_INNER
#define DS   16       // D_STATE
#define DTR  12       // DT_RANK
#define BATCH 4
#define SEQ  3136
#define BLR  (BATCH*SEQ)   // 12544 rows
#define NPROJ 768          // 2*D_INNER
#define XPN  48            // padded DT_RANK+2*D_STATE (44 -> 48)
#define NCH  32            // scan chunks
#define CHL  (SEQ/NCH)     // 98

typedef _Float16 half8  __attribute__((ext_vector_type(8)));
typedef _Float16 half16 __attribute__((ext_vector_type(16)));
typedef float    f32x8  __attribute__((ext_vector_type(8)));

// ---------------------------------------------------------------- weights prep
__global__ void prep_weights(const float* __restrict__ inw,   // (192,768)
                             const float* __restrict__ xpw,   // (384,44)
                             const float* __restrict__ outw,  // (384,192)
                             _Float16* __restrict__ wt1,      // (768,192)
                             _Float16* __restrict__ wt2,      // (48,384)
                             _Float16* __restrict__ wt3)      // (192,384)
{
    const int N1 = NPROJ * DM;   // 147456
    const int N2 = XPN * DI;     // 18432
    const int N3 = DM * DI;      // 73728
    int i = blockIdx.x * blockDim.x + threadIdx.x;
    if (i < N1) {
        int n = i / DM, k = i % DM;
        wt1[i] = (_Float16)inw[(size_t)k * NPROJ + n];
    } else if (i < N1 + N2) {
        int j = i - N1; int n = j / DI, k = j % DI;
        float v = (n < DTR + 2 * DS) ? xpw[(size_t)k * (DTR + 2 * DS) + n] : 0.f;
        wt2[j] = (_Float16)v;
    } else if (i < N1 + N2 + N3) {
        int j = i - N1 - N2; int n = j / DI, k = j % DI;
        wt3[j] = (_Float16)outw[(size_t)k * DM + n];
    }
}

// ---------------------------------------------------------------- layernorm -> f16
__global__ void ln_kernel(const float* __restrict__ x, const float* __restrict__ w,
                          const float* __restrict__ b, _Float16* __restrict__ h)
{
    int wave = (blockIdx.x * blockDim.x + threadIdx.x) >> 5;   // one wave per row
    int lane = threadIdx.x & 31;
    const float* xr = x + (size_t)wave * DM;
    float v[6], s = 0.f, sq = 0.f;
#pragma unroll
    for (int i = 0; i < 6; ++i) { v[i] = xr[lane + 32 * i]; s += v[i]; sq += v[i] * v[i]; }
#pragma unroll
    for (int off = 16; off > 0; off >>= 1) {
        s  += __shfl_xor(s,  off, 32);
        sq += __shfl_xor(sq, off, 32);
    }
    float mu = s * (1.f / DM);
    float var = sq * (1.f / DM) - mu * mu;
    float rs = rsqrtf(var + 1e-5f);
#pragma unroll
    for (int i = 0; i < 6; ++i) {
        int c = lane + 32 * i;
        h[(size_t)wave * DM + c] = (_Float16)((v[i] - mu) * rs * w[c] + b[c]);
    }
}

// ---------------------------------------------------------------- WMMA GEMM
// Each wave computes 1 M-tile x NT N-tiles (16 x 16*NT output), reusing the A
// fragment across NT WMMAs per k-step.
// A: row-major (M x K) f16.  Bt: N-major (N x K) f16.  K multiple of 32.
template <int EPI, int NT>
__global__ void gemm_wmma_kernel(const _Float16* __restrict__ A,
                                 const _Float16* __restrict__ Bt,
                                 int K, int tilesN,
                                 float* __restrict__ o32,
                                 _Float16* __restrict__ o16)
{
    int lane = threadIdx.x & 31;
    int group = blockIdx.x * (blockDim.x >> 5) + (threadIdx.x >> 5);
    int groupsN = tilesN / NT;
    int tm  = group / groupsN;
    int tn0 = (group % groupsN) * NT;

    // A fragment: lane<16 holds row m, K {0..7,16..23}; lane>=16 K {8..15,24..31}
    int rowA  = tm * 16 + (lane & 15);
    int koffA = (lane & 16) ? 8 : 0;
    const _Float16* pa = A + (size_t)rowA * K + koffA;
    // B fragment: lane holds col n; lane<16 K {0..7,16..23}, lane>=16 K {8..15,24..31}
    int nB    = lane & 15;
    int koffB = (lane & 16) ? 8 : 0;
    const _Float16* pb[NT];
#pragma unroll
    for (int t = 0; t < NT; ++t)
        pb[t] = Bt + (size_t)((tn0 + t) * 16 + nB) * K + koffB;

    f32x8 acc[NT];
#pragma unroll
    for (int t = 0; t < NT; ++t)
        acc[t] = (f32x8){0.f, 0.f, 0.f, 0.f, 0.f, 0.f, 0.f, 0.f};

    for (int k0 = 0; k0 < K; k0 += 32) {
        half8 alo = *(const half8*)(pa + k0);
        half8 ahi = *(const half8*)(pa + k0 + 16);
        half16 av;
#pragma unroll
        for (int i = 0; i < 8; ++i) { av[i] = alo[i]; av[i + 8] = ahi[i]; }
#pragma unroll
        for (int t = 0; t < NT; ++t) {
            half8 blo = *(const half8*)(pb[t] + k0);
            half8 bhi = *(const half8*)(pb[t] + k0 + 16);
            half16 bv;
#pragma unroll
            for (int i = 0; i < 8; ++i) { bv[i] = blo[i]; bv[i + 8] = bhi[i]; }
            acc[t] = __builtin_amdgcn_wmma_f32_16x16x32_f16(false, av, false, bv,
                                                            (short)0, acc[t], false, false);
        }
    }

    // C/D layout: VGPR r -> row r (+8 for upper lane half), col = lane&15
    int rbase = tm * 16 + ((lane & 16) ? 8 : 0);
#pragma unroll
    for (int t = 0; t < NT; ++t) {
        int col = (tn0 + t) * 16 + (lane & 15);
#pragma unroll
        for (int r = 0; r < 8; ++r) {
            int row = rbase + r;
            float c = acc[t][r];
            if (EPI == 0) {                       // in_proj: u_pre f32 | silu(z) f16
                if (col < DI) o32[(size_t)row * DI + col] = c;
                else {
                    float s = c / (1.f + __expf(-c));
                    o16[(size_t)row * DI + (col - DI)] = (_Float16)s;
                }
            } else if (EPI == 1) {                // x_proj -> x_dbl (ld 48)
                o32[(size_t)row * XPN + col] = c;
            } else {                              // out_proj -> d_out (ld 192)
                o32[(size_t)row * DM + col] = c;
            }
        }
    }
}

// ---------------------------------------------------------------- causal depthwise conv4 + SiLU
__global__ void conv_silu(const float* __restrict__ up, const float* __restrict__ cw,
                          const float* __restrict__ cb, float* __restrict__ u32,
                          _Float16* __restrict__ u16)
{
    int i = blockIdx.x * blockDim.x + threadIdx.x;   // BLR*DI
    int d = i % DI;
    int bl = i / DI;
    int b = bl / SEQ, l = bl % SEQ;
    float acc = cb[d];
#pragma unroll
    for (int j = 0; j < 4; ++j) {
        int ll = l - 3 + j;
        if (ll >= 0) acc += up[((size_t)(b * SEQ + ll)) * DI + d] * cw[d * 4 + j];
    }
    float s = acc / (1.f + __expf(-acc));
    u32[i] = s;
    u16[i] = (_Float16)s;
}

// ---------------------------------------------------------------- dt_proj (K=12) + softplus
__global__ void dtproj_kernel(const float* __restrict__ xd, const float* __restrict__ dw,
                              const float* __restrict__ db, float* __restrict__ delta)
{
    int i = blockIdx.x * blockDim.x + threadIdx.x;   // BLR*DI
    int d = i % DI;
    int bl = i / DI;
    float acc = db[d];
#pragma unroll
    for (int r = 0; r < DTR; ++r)
        acc += xd[(size_t)bl * XPN + r] * dw[r * DI + d];
    delta[i] = (acc > 20.f) ? acc : log1pf(__expf(acc));
}

// ---------------------------------------------------------------- scan pass 1: per-chunk S and P
__global__ void scan_pass1(const float* __restrict__ delta, const float* __restrict__ u,
                           const float* __restrict__ xd, const float* __restrict__ A_log,
                           float* __restrict__ Sc, float* __restrict__ Pc)
{
    int d = threadIdx.x;           // 0..383
    int c = blockIdx.x;            // chunk
    int b = blockIdx.y;            // batch
    float Am[DS];
#pragma unroll
    for (int n = 0; n < DS; ++n) Am[n] = -__expf(A_log[d * DS + n]);
    float h[DS];
#pragma unroll
    for (int n = 0; n < DS; ++n) h[n] = 0.f;
    float sd = 0.f;
    int l0 = c * CHL;
    for (int l = l0; l < l0 + CHL; ++l) {
        size_t bl = (size_t)b * SEQ + l;
        float dl = delta[bl * DI + d];
        float du = dl * u[bl * DI + d];
        sd += dl;
        const float4* bp = (const float4*)(xd + bl * XPN + DTR);
        float4 B0 = bp[0], B1 = bp[1], B2 = bp[2], B3 = bp[3];
        float Bv[DS] = {B0.x, B0.y, B0.z, B0.w, B1.x, B1.y, B1.z, B1.w,
                        B2.x, B2.y, B2.z, B2.w, B3.x, B3.y, B3.z, B3.w};
#pragma unroll
        for (int n = 0; n < DS; ++n) {
            float dA = __expf(dl * Am[n]);
            h[n] = dA * h[n] + du * Bv[n];
        }
    }
    size_t base = (((size_t)b * NCH + c) * DI + d) * DS;
#pragma unroll
    for (int n = 0; n < DS; ++n) {
        Sc[base + n] = h[n];
        Pc[base + n] = __expf(sd * Am[n]);   // prod of exp(dl*A) == exp(A * sum dl)
    }
}

// ---------------------------------------------------------------- sequential chunk combine
__global__ void scan_combine(const float* __restrict__ Sc, const float* __restrict__ Pc,
                             float* __restrict__ Hin)
{
    int i = blockIdx.x * blockDim.x + threadIdx.x;   // BATCH*DI*DS
    int n = i % DS;
    int d = (i / DS) % DI;
    int b = i / (DS * DI);
    float h = 0.f;
    for (int c = 0; c < NCH; ++c) {
        size_t idx = (((size_t)b * NCH + c) * DI + d) * DS + n;
        Hin[idx] = h;
        h = Pc[idx] * h + Sc[idx];
    }
}

// ---------------------------------------------------------------- scan pass 2: replay + gate
__global__ void scan_pass2(const float* __restrict__ delta, const float* __restrict__ u,
                           const float* __restrict__ xd, const float* __restrict__ A_log,
                           const float* __restrict__ Hin, const float* __restrict__ Dskip,
                           const _Float16* __restrict__ zs, _Float16* __restrict__ yg)
{
    int d = threadIdx.x;
    int c = blockIdx.x;
    int b = blockIdx.y;
    float Am[DS];
#pragma unroll
    for (int n = 0; n < DS; ++n) Am[n] = -__expf(A_log[d * DS + n]);
    float h[DS];
    size_t hb = (((size_t)b * NCH + c) * DI + d) * DS;
#pragma unroll
    for (int n = 0; n < DS; ++n) h[n] = Hin[hb + n];
    float dsk = Dskip[d];
    int l0 = c * CHL;
    for (int l = l0; l < l0 + CHL; ++l) {
        size_t bl = (size_t)b * SEQ + l;
        float dl = delta[bl * DI + d];
        float uu = u[bl * DI + d];
        float du = dl * uu;
        const float4* bp = (const float4*)(xd + bl * XPN + DTR);
        const float4* cp = (const float4*)(xd + bl * XPN + DTR + DS);
        float4 B0 = bp[0], B1 = bp[1], B2 = bp[2], B3 = bp[3];
        float4 C0 = cp[0], C1 = cp[1], C2 = cp[2], C3 = cp[3];
        float Bv[DS] = {B0.x, B0.y, B0.z, B0.w, B1.x, B1.y, B1.z, B1.w,
                        B2.x, B2.y, B2.z, B2.w, B3.x, B3.y, B3.z, B3.w};
        float Cv[DS] = {C0.x, C0.y, C0.z, C0.w, C1.x, C1.y, C1.z, C1.w,
                        C2.x, C2.y, C2.z, C2.w, C3.x, C3.y, C3.z, C3.w};
        float y = 0.f;
#pragma unroll
        for (int n = 0; n < DS; ++n) {
            float dA = __expf(dl * Am[n]);
            h[n] = dA * h[n] + du * Bv[n];
            y += h[n] * Cv[n];
        }
        y += uu * dsk;
        float g = (float)zs[bl * DI + d];
        yg[bl * DI + d] = (_Float16)(y * g);
    }
}

// ================================================================ launch
extern "C" void kernel_launch(void* const* d_in, const int* in_sizes, int n_in,
                              void* d_out, int out_size, void* d_ws, size_t ws_size,
                              hipStream_t stream)
{
    const float* x        = (const float*)d_in[0];
    const float* ln_w     = (const float*)d_in[1];
    const float* ln_b     = (const float*)d_in[2];
    const float* in_projw = (const float*)d_in[3];
    const float* conv_w   = (const float*)d_in[4];
    const float* conv_b   = (const float*)d_in[5];
    const float* x_projw  = (const float*)d_in[6];
    const float* dt_projw = (const float*)d_in[7];
    const float* dt_projb = (const float*)d_in[8];
    const float* A_log    = (const float*)d_in[9];
    const float* D_skip   = (const float*)d_in[10];
    const float* out_projw= (const float*)d_in[11];
    float* out            = (float*)d_out;

    char* p = (char*)d_ws;
    size_t off = 0;
    auto alloc = [&](size_t bytes) -> char* {
        char* r = p + off;
        off += (bytes + 255) & ~(size_t)255;
        return r;
    };
    _Float16* h16  = (_Float16*)alloc((size_t)BLR * DM * 2);
    _Float16* wt1  = (_Float16*)alloc((size_t)NPROJ * DM * 2);
    _Float16* wt2  = (_Float16*)alloc((size_t)XPN * DI * 2);
    _Float16* wt3  = (_Float16*)alloc((size_t)DM * DI * 2);
    float*    upre = (float*)   alloc((size_t)BLR * DI * 4);
    _Float16* zs   = (_Float16*)alloc((size_t)BLR * DI * 2);
    float*    u32  = (float*)   alloc((size_t)BLR * DI * 4);
    _Float16* u16  = (_Float16*)alloc((size_t)BLR * DI * 2);
    float*    xdbl = (float*)   alloc((size_t)BLR * XPN * 4);
    float*    dlt  = (float*)   alloc((size_t)BLR * DI * 4);
    float*    Sc   = (float*)   alloc((size_t)BATCH * NCH * DI * DS * 4);
    float*    Pc   = (float*)   alloc((size_t)BATCH * NCH * DI * DS * 4);
    float*    Hin  = (float*)   alloc((size_t)BATCH * NCH * DI * DS * 4);
    _Float16* yg   = (_Float16*)alloc((size_t)BLR * DI * 2);
    (void)ws_size; (void)in_sizes; (void)n_in; (void)out_size;

    const int prepN = NPROJ * DM + XPN * DI + DM * DI;   // 239616
    prep_weights<<<(prepN + 255) / 256, 256, 0, stream>>>(in_projw, x_projw, out_projw,
                                                          wt1, wt2, wt3);

    ln_kernel<<<BLR / 8, 256, 0, stream>>>(x, ln_w, ln_b, h16);

    // in_proj: M=12544, N=768, K=192 -> 784 * (48/4) groups, 8 waves/block
    gemm_wmma_kernel<0, 4><<<(BLR / 16) * (NPROJ / 64) / 8, 256, 0, stream>>>(
        h16, wt1, DM, NPROJ / 16, upre, zs);

    conv_silu<<<(BLR * DI) / 256, 256, 0, stream>>>(upre, conv_w, conv_b, u32, u16);

    // x_proj: M=12544, N=48, K=384 -> 784 * 1 groups (NT=3)
    gemm_wmma_kernel<1, 3><<<(BLR / 16) / 8, 256, 0, stream>>>(
        u16, wt2, DI, XPN / 16, xdbl, nullptr);

    dtproj_kernel<<<(BLR * DI) / 256, 256, 0, stream>>>(xdbl, dt_projw, dt_projb, dlt);

    dim3 scangrid(NCH, BATCH);
    scan_pass1<<<scangrid, DI, 0, stream>>>(dlt, u32, xdbl, A_log, Sc, Pc);
    scan_combine<<<(BATCH * DI * DS) / 256, 256, 0, stream>>>(Sc, Pc, Hin);
    scan_pass2<<<scangrid, DI, 0, stream>>>(dlt, u32, xdbl, A_log, Hin, D_skip, zs, yg);

    // out_proj: M=12544, N=192, K=384 -> 784 * 3 groups (NT=4)
    gemm_wmma_kernel<2, 4><<<(BLR / 16) * (DM / 64) / 8, 256, 0, stream>>>(
        yg, wt3, DI, DM / 16, out, nullptr);
}
